// FeedForward_62818191671512
// MI455X (gfx1250) — matrix operand order repaired and checked by reading, not compiled, not run on hardware
//
#include <hip/hip_runtime.h>
#include <hip/hip_bf16.h>

// MoE SwiGLU FFN for MI455X (gfx1250), bf16 WMMA with fp32 accumulate.
// T=8192 tokens, DIM=1024, HIDDEN=4096, 8 experts, top-2 routing.
//
// Pipeline: [memset] -> router (top-2 softmax, expert bucketing)
//           -> [f32->bf16 weight/x conversion, if ws fits]
//           -> fused per-expert SwiGLU FFN (WMMA bf16) with atomic scatter.

#define DIM 1024
#define HIDDEN 4096
#define NE 8
#define T_TOK 8192
#define TILE_M 32
#define KH 128                // hidden chunk per outer iteration (8 waves x 16)
#define XS_STRIDE (DIM + 8)   // bf16 elements; +8 spreads LDS banks
#define HS_STRIDE (KH + 8)

typedef __attribute__((ext_vector_type(16))) __bf16 v16bf;
typedef __attribute__((ext_vector_type(8)))  __bf16 v8bf;
typedef __attribute__((ext_vector_type(8)))  float  v8f;

// Native f32->bf16 (backend emits v_cvt_pk_bf16_f32 on gfx1250, RNE).
__device__ __forceinline__ __bf16 f2bf(float f) { return (__bf16)f; }

// Load a 16-element B fragment (values p[0..7] and p[16..23] of one matrix row).
template <typename DT>
__device__ __forceinline__ v16bf loadB(const DT* __restrict__ p) {
  if constexpr (sizeof(DT) == 2) {
    const v8bf* q = (const v8bf*)p;
    v8bf lo = q[0], hi = q[2];
    return __builtin_shufflevector(lo, hi, 0, 1, 2, 3, 4, 5, 6, 7, 8, 9, 10,
                                   11, 12, 13, 14, 15);
  } else {
    float4 a = ((const float4*)p)[0];
    float4 b = ((const float4*)p)[1];
    float4 c = ((const float4*)p)[4];
    float4 d = ((const float4*)p)[5];
    v16bf r;
    r[0]  = f2bf(a.x); r[1]  = f2bf(a.y); r[2]  = f2bf(a.z); r[3]  = f2bf(a.w);
    r[4]  = f2bf(b.x); r[5]  = f2bf(b.y); r[6]  = f2bf(b.z); r[7]  = f2bf(b.w);
    r[8]  = f2bf(c.x); r[9]  = f2bf(c.y); r[10] = f2bf(c.z); r[11] = f2bf(c.w);
    r[12] = f2bf(d.x); r[13] = f2bf(d.y); r[14] = f2bf(d.z); r[15] = f2bf(d.w);
    return r;
  }
}

// A-fragment from LDS: lane holds row M=lane&15; K halves selected by lane>>4
// (matches 16-bit A 16x32 VGPR layout, ISA 7.12.2).
__device__ __forceinline__ v16bf load_a(const __bf16* p) {
  v8bf lo = *(const v8bf*)p;
  v8bf hi = *(const v8bf*)(p + 16);
  return __builtin_shufflevector(lo, hi, 0, 1, 2, 3, 4, 5, 6, 7, 8, 9, 10, 11,
                                 12, 13, 14, 15);
}

#define WMMA_BF16(A, B, C) \
  __builtin_amdgcn_wmma_f32_16x16x32_bf16(false, (A), false, (B), (short)0, (C), false, false)

// ---------------- One-time f32 -> bf16 conversion (streaming) ---------------
__global__ __launch_bounds__(256, 1)
void cvt_bf16_kernel(const float* __restrict__ src, __bf16* __restrict__ dst,
                     long long n) {
  long long i = ((long long)blockIdx.x * 256 + threadIdx.x) * 8;
  if (i + 8 > n) return;   // all sizes are multiples of 2048, exact fit
  float4 a = *(const float4*)(src + i);
  float4 b = *(const float4*)(src + i + 4);
  v8bf r;
  r[0] = f2bf(a.x); r[1] = f2bf(a.y); r[2] = f2bf(a.z); r[3] = f2bf(a.w);
  r[4] = f2bf(b.x); r[5] = f2bf(b.y); r[6] = f2bf(b.z); r[7] = f2bf(b.w);
  *(v8bf*)(dst + i) = r;
}

// ---------------- Router: top-2 softmax gating + expert bucketing -----------
__global__ __launch_bounds__(256, 1)
void router_kernel(const float* __restrict__ x, const float* __restrict__ gw,
                   int* __restrict__ counts, int* __restrict__ tokid,
                   float* __restrict__ tokw) {
  const int lane = threadIdx.x & 31;
  const int wv   = threadIdx.x >> 5;
  const int t    = blockIdx.x * 8 + wv;

  float acc[NE];
#pragma unroll
  for (int e = 0; e < NE; e++) acc[e] = 0.f;

  const float* xr = x + (size_t)t * DIM;
  for (int k = lane; k < DIM; k += 32) {
    float xv = xr[k];
#pragma unroll
    for (int e = 0; e < NE; e++) acc[e] += xv * gw[e * DIM + k];
  }
#pragma unroll
  for (int off = 16; off > 0; off >>= 1)
#pragma unroll
    for (int e = 0; e < NE; e++) acc[e] += __shfl_xor(acc[e], off, 32);

  if (lane == 0) {
    int   e0 = 0; float b0 = acc[0];
#pragma unroll
    for (int e = 1; e < NE; e++) if (acc[e] > b0) { b0 = acc[e]; e0 = e; }
    int   e1 = (e0 == 0) ? 1 : 0; float b1 = acc[e1];
#pragma unroll
    for (int e = 0; e < NE; e++)
      if (e != e0 && acc[e] > b1) { b1 = acc[e]; e1 = e; }
    // renormalized top-2 softmax weights (shared Z cancels)
    float p1 = __expf(b1 - b0);
    float inv = 1.f / (1.f + p1);
    float w0 = inv, w1 = p1 * inv;

    int pos0 = atomicAdd(&counts[e0], 1);
    tokid[e0 * T_TOK + pos0] = t;
    tokw [e0 * T_TOK + pos0] = w0;
    int pos1 = atomicAdd(&counts[e1], 1);
    tokid[e1 * T_TOK + pos1] = t;
    tokw [e1 * T_TOK + pos1] = w1;
  }
}

// ---------------- Fused SwiGLU expert FFN (per 32-token tile) ---------------
template <typename DT>   // DT = __bf16 (preconverted workspace) or float
__global__ __launch_bounds__(256, 1)
void moe_ffn_kernel(const DT* __restrict__ x,
                    const DT* __restrict__ w1,
                    const DT* __restrict__ w2,
                    const DT* __restrict__ w3,
                    const int*   __restrict__ counts,
                    const int*   __restrict__ tokid,
                    const float* __restrict__ tokw,
                    float* __restrict__ out) {
  const int ex    = blockIdx.y;
  const int mtile = blockIdx.x;
  const int cnt   = counts[ex];
  if (mtile * TILE_M >= cnt) return;       // block-uniform early exit

  __shared__ __bf16 xs[TILE_M][XS_STRIDE]; // gathered x tile (bf16)
  __shared__ __bf16 hs[TILE_M][HS_STRIDE]; // weighted silu(xW1)*(xW3) chunk
  __shared__ int    toks[TILE_M];
  __shared__ float  wts[TILE_M];

  const int tid  = threadIdx.x;
  const int lane = tid & 31;
  const int wv   = tid >> 5;          // 8 waves
  const int arow = lane & 15;         // M index / N index within 16-tile
  const int ksel = lane >> 4;         // which K half this lane holds

  if (tid < TILE_M) {
    int r   = mtile * TILE_M + tid;
    int idx = (r < cnt) ? r : (cnt - 1);
    toks[tid] = tokid[ex * T_TOK + idx];
    wts[tid]  = (r < cnt) ? tokw[ex * T_TOK + idx] : 0.0f; // pad rows -> 0
  }
  __syncthreads();

  { // gather x tile -> LDS; 8 threads per row
    int row = tid >> 3;
    int cb  = (tid & 7) * 128;
    const DT* src = x + (size_t)toks[row] * DIM + cb;
    if constexpr (sizeof(DT) == 4) {
#pragma unroll 8
      for (int c = 0; c < 128; c += 4) {
        float4 v = *(const float4*)(src + c);
        xs[row][cb + c + 0] = f2bf(v.x);
        xs[row][cb + c + 1] = f2bf(v.y);
        xs[row][cb + c + 2] = f2bf(v.z);
        xs[row][cb + c + 3] = f2bf(v.w);
      }
    } else {
#pragma unroll
      for (int c = 0; c < 128; c += 8)
        *(v8bf*)&xs[row][cb + c] = *(const v8bf*)(src + c);
    }
  }
  __syncthreads();

  v8f oacc[16] = {};                  // 32 x 128 fp32 output accum per wave

  const __bf16* pa0 = &xs[arow][ksel * 8];
  const __bf16* pa1 = &xs[16 + arow][ksel * 8];

  for (int kh = 0; kh < HIDDEN; kh += KH) {
    // ---- stage 1: h = wt * silu(x@w1^T) * (x@w3^T) for 16 hidden cols/wave
    v8f s0 = {}, s1 = {}, g0 = {}, g1 = {};
    const int hrow = kh + wv * 16 + arow;   // B lane = hidden unit (column N)
    const DT* pw1 = w1 + ((size_t)ex * HIDDEN + hrow) * DIM + ksel * 8;
    const DT* pw3 = w3 + ((size_t)ex * HIDDEN + hrow) * DIM + ksel * 8;

    for (int ks = 0; ks < DIM; ks += 32) {
      v16bf a0 = load_a(pa0 + ks);
      v16bf a1 = load_a(pa1 + ks);
      v16bf b1 = loadB(pw1 + ks);
      v16bf b3 = loadB(pw3 + ks);
      s0 = WMMA_BF16(a0, b1, s0);
      s1 = WMMA_BF16(a1, b1, s1);
      g0 = WMMA_BF16(a0, b3, g0);
      g1 = WMMA_BF16(a1, b3, g1);
    }

    const int hcol  = wv * 16 + arow;
    const int mbase = ksel * 8;
#pragma unroll
    for (int r = 0; r < 8; r++) {
      float sv = s0[r];
      float hv = (sv / (1.f + __expf(-sv))) * g0[r] * wts[mbase + r];
      hs[mbase + r][hcol] = f2bf(hv);
      float sv1 = s1[r];
      float hv1 = (sv1 / (1.f + __expf(-sv1))) * g1[r] * wts[16 + mbase + r];
      hs[16 + mbase + r][hcol] = f2bf(hv1);
    }
    __syncthreads();

    // ---- stage 2: out_acc += h_chunk @ w2_chunk^T (wave owns 128 out dims)
    const __bf16* ha0 = &hs[arow][ksel * 8];
    const __bf16* ha1 = &hs[16 + arow][ksel * 8];
#pragma unroll
    for (int ks = 0; ks < KH; ks += 32) {
      v16bf a0 = load_a(ha0 + ks);
      v16bf a1 = load_a(ha1 + ks);
#pragma unroll
      for (int nt = 0; nt < 8; nt++) {
        const int orow = wv * 128 + nt * 16 + arow;  // output dim (column N)
        const DT* pw2 =
            w2 + ((size_t)ex * DIM + orow) * HIDDEN + kh + ks + ksel * 8;
        v16bf b = loadB(pw2);
        oacc[nt * 2 + 0] = WMMA_BF16(a0, b, oacc[nt * 2 + 0]);
        oacc[nt * 2 + 1] = WMMA_BF16(a1, b, oacc[nt * 2 + 1]);
      }
    }
    __syncthreads();   // hs reused next chunk
  }

  // ---- epilogue: scatter-add (token routed to 2 experts -> atomics)
  const int ncol = wv * 128;
#pragma unroll
  for (int mt = 0; mt < 2; mt++) {
#pragma unroll
    for (int nt = 0; nt < 8; nt++) {
      v8f a = oacc[nt * 2 + mt];
#pragma unroll
      for (int r = 0; r < 8; r++) {
        int m = mt * 16 + ksel * 8 + r;
        if (mtile * TILE_M + m < cnt) {
          float* p = out + (size_t)toks[m] * DIM + ncol + nt * 16 + arow;
          __hip_atomic_fetch_add(p, a[r], __ATOMIC_RELAXED,
                                 __HIP_MEMORY_SCOPE_AGENT);
        }
      }
    }
  }
}

extern "C" void kernel_launch(void* const* d_in, const int* in_sizes, int n_in,
                              void* d_out, int out_size, void* d_ws,
                              size_t ws_size, hipStream_t stream) {
  const float* x      = (const float*)d_in[0];
  const float* gate_w = (const float*)d_in[1];
  const float* w1     = (const float*)d_in[2];
  const float* w2     = (const float*)d_in[3];
  const float* w3     = (const float*)d_in[4];
  float* out = (float*)d_out;

  const size_t N_X = (size_t)T_TOK * DIM;          // 8.39M
  const size_t N_W = (size_t)NE * HIDDEN * DIM;    // 33.55M (each of w1/w2/w3)

  char* ws = (char*)d_ws;
  size_t off = 0;
  int*   counts = (int*)(ws + off);  off += 256;
  int*   tokid  = (int*)(ws + off);  off += (size_t)NE * T_TOK * 4;
  float* tokw   = (float*)(ws + off); off += (size_t)NE * T_TOK * 4;
  __bf16* xbf  = (__bf16*)(ws + off); off += N_X * 2;
  __bf16* w1bf = (__bf16*)(ws + off); off += N_W * 2;
  __bf16* w3bf = (__bf16*)(ws + off); off += N_W * 2;
  __bf16* w2bf = (__bf16*)(ws + off); off += N_W * 2;
  const size_t NEED = off;

  hipMemsetAsync(counts, 0, 256, stream);
  hipMemsetAsync(d_out, 0, (size_t)out_size * sizeof(float), stream);

  router_kernel<<<T_TOK / 8, 256, 0, stream>>>(x, gate_w, counts, tokid, tokw);

  dim3 grid(T_TOK / TILE_M, NE);
  if (ws_size >= NEED) {
    // One-time streaming conversion: halves weight bytes in the GEMM and
    // removes all cvt VALU work from the WMMA hot loop.
    cvt_bf16_kernel<<<(int)(N_X / 2048), 256, 0, stream>>>(x, xbf, (long long)N_X);
    cvt_bf16_kernel<<<(int)(N_W / 2048), 256, 0, stream>>>(w1, w1bf, (long long)N_W);
    cvt_bf16_kernel<<<(int)(N_W / 2048), 256, 0, stream>>>(w3, w3bf, (long long)N_W);
    cvt_bf16_kernel<<<(int)(N_W / 2048), 256, 0, stream>>>(w2, w2bf, (long long)N_W);
    moe_ffn_kernel<__bf16><<<grid, 256, 0, stream>>>(xbf, w1bf, w2bf, w3bf,
                                                     counts, tokid, tokw, out);
  } else {
    moe_ffn_kernel<float><<<grid, 256, 0, stream>>>(x, w1, w2, w3,
                                                    counts, tokid, tokw, out);
  }
}
